// MicroBlock_15272903704777
// MI455X (gfx1250) — matrix operand order repaired and checked by reading, not compiled, not run on hardware
//
#include <hip/hip_runtime.h>

// ---------------------------------------------------------------------------
// Fused ShuffleNet-style microblock for MI455X (gfx1250, wave32, WMMA).
// ---------------------------------------------------------------------------

typedef __attribute__((ext_vector_type(16))) _Float16 v16h;
typedef __attribute__((ext_vector_type(8)))  _Float16 v8h;
typedef __attribute__((ext_vector_type(8)))  float    v8f;

namespace {
constexpr int kB  = 16;
constexpr int kC  = 96;      // CIN
constexpr int kH  = 112;
constexpr int kW  = 112;
constexpr int kHW = kH * kW; // 12544
constexpr int kC1 = 192;
constexpr int kC2 = 384;
constexpr int kSQ = 24;
constexpr int kKS = 392;     // padded LDS row stride (halves) for act panel
constexpr float kEPS = 1e-5f;

// workspace byte offsets
constexpr size_t WP_OFF  = 0;      // f16 pw weights in A-operand lane layout: [6 mt][12 ks][32 lane][16 half] = 73728 B
constexpr size_t WD1_OFF = 73728;  // 192*3 f32 folded dw1 taps
constexpr size_t O1_OFF  = 76032;  // 192 f32 bn1 bias
constexpr size_t WD2_OFF = 76800;  // 384*3 f32 folded dw2 taps
constexpr size_t O2_OFF  = 81408;  // 384 f32 bn2 bias
constexpr size_t S3_OFF  = 82944;  // 96 f32 bn3 scale
constexpr size_t O3_OFF  = 83328;  // 96 f32 bn3 bias
constexpr size_t SUM_OFF = 83712;  // 16*96 f32 channel sums
constexpr size_t A1_OFF  = 89856;  // 16*96 f32
constexpr size_t B1_OFF  = 96000;  // 16*96 f32

// LDS layout (bytes) inside fused kernel
constexpr int LDS_T1   = 0;                       // 192*112 f16 = 43008 B
constexpr int LDS_ACT  = 43008;                   // 112*392 f16 = 87808 B
constexpr int LDS_SUM  = 43008 + 87808;           // 96 f32
constexpr int LDS_SIZE = LDS_SUM + 96 * 4;        // 131200 B
} // namespace

// ---------------------------------------------------------------------------
// Kernel 0: fold BN into dw taps, build WMMA-A-layout f16 pw weights,
// zero the channel-sum accumulators.
// ---------------------------------------------------------------------------
__global__ void prep_kernel(const float* __restrict__ wdw1,
                            const float* __restrict__ g1, const float* __restrict__ bb1,
                            const float* __restrict__ m1, const float* __restrict__ vv1,
                            const float* __restrict__ wdw2,
                            const float* __restrict__ g2, const float* __restrict__ bb2,
                            const float* __restrict__ m2, const float* __restrict__ vv2,
                            const float* __restrict__ wpw,
                            const float* __restrict__ g3, const float* __restrict__ bb3,
                            const float* __restrict__ m3, const float* __restrict__ vv3,
                            char* __restrict__ ws) {
  const int tid = blockIdx.x * blockDim.x + threadIdx.x;
  const int nth = gridDim.x * blockDim.x;

  float*     wd1f = (float*)(ws + WD1_OFF);
  float*     o1   = (float*)(ws + O1_OFF);
  float*     wd2f = (float*)(ws + WD2_OFF);
  float*     o2   = (float*)(ws + O2_OFF);
  float*     s3   = (float*)(ws + S3_OFF);
  float*     o3   = (float*)(ws + O3_OFF);
  float*     sums = (float*)(ws + SUM_OFF);
  _Float16*  wp   = (_Float16*)(ws + WP_OFF);

  for (int i = tid; i < kB * kC; i += nth) sums[i] = 0.0f;

  for (int c = tid; c < kC1; c += nth) {
    float s = g1[c] * rsqrtf(vv1[c] + kEPS);
    o1[c] = bb1[c] - m1[c] * s;
    wd1f[c * 3 + 0] = wdw1[c * 3 + 0] * s;
    wd1f[c * 3 + 1] = wdw1[c * 3 + 1] * s;
    wd1f[c * 3 + 2] = wdw1[c * 3 + 2] * s;
  }
  for (int c = tid; c < kC2; c += nth) {
    float s = g2[c] * rsqrtf(vv2[c] + kEPS);
    o2[c] = bb2[c] - m2[c] * s;
    wd2f[c * 3 + 0] = wdw2[c * 3 + 0] * s;
    wd2f[c * 3 + 1] = wdw2[c * 3 + 1] * s;
    wd2f[c * 3 + 2] = wdw2[c * 3 + 2] * s;
  }
  for (int c = tid; c < kC; c += nth) {
    float s = g3[c] * rsqrtf(vv3[c] + kEPS);
    s3[c] = s;
    o3[c] = bb3[c] - m3[c] * s;
  }

  // A-operand (16x32 f16) per-lane layout, ISA 7.12.2:
  //   lanes 0-15  hold K = {0..7, 16..23},  lanes 16-31 hold K = {8..15, 24..31}
  // half index h -> koff = h + 8*lo + (h>=8 ? 8 : 0)
  for (int i = tid; i < 6 * 12 * 32 * 16; i += nth) {
    int hx   = i & 15;
    int lane = (i >> 4) & 31;
    int t    = i >> 9;
    int ks   = t % 12;
    int mt   = t / 12;
    int lo   = lane >> 4;
    int co   = mt * 16 + (lane & 15);
    int koff = hx + 8 * lo + (hx >= 8 ? 8 : 0);
    int k    = ks * 32 + koff;           // activations are stored pre-shuffled, so k-order is natural
    wp[i] = (_Float16)wpw[co * kC2 + k];
  }
}

// ---------------------------------------------------------------------------
// Kernel 1: per (batch, row): dw1(3x1,H) + bn1 -> LDS f16
//                              dw2(1x3,W) + bn2 + relu6 + shuffle(6) -> LDS f16 panel
//                              pointwise 96x384 GEMM via v_wmma_f32_16x16x32_f16
//                              bn3 epilogue, write h to d_out, accumulate channel sums
// ---------------------------------------------------------------------------
__global__ void fused_main_kernel(const float* __restrict__ x,
                                  float* __restrict__ out,
                                  char* __restrict__ ws) {
  extern __shared__ char smem[];
  _Float16* t1  = (_Float16*)(smem + LDS_T1);    // [192][112]
  _Float16* act = (_Float16*)(smem + LDS_ACT);   // [112][kKS], shuffled channel order
  float*    bsm = (float*)(smem + LDS_SUM);      // [96]

  const int b  = blockIdx.x / kH;
  const int hh = blockIdx.x % kH;
  const int tid = threadIdx.x;

  const float* wd1f = (const float*)(ws + WD1_OFF);
  const float* o1   = (const float*)(ws + O1_OFF);
  const float* wd2f = (const float*)(ws + WD2_OFF);
  const float* o2   = (const float*)(ws + O2_OFF);
  const float* s3   = (const float*)(ws + S3_OFF);
  const float* o3   = (const float*)(ws + O3_OFF);
  const _Float16* wp = (const _Float16*)(ws + WP_OFF);
  float* gsums = (float*)(ws + SUM_OFF);

  if (tid < kC) bsm[tid] = 0.0f;

  // ---- stage 1: t1 = bn1(dw1(x)) along H (3x1, pad 1) -------------------
  const float* xb = x + (size_t)b * kC * kHW;
  for (int i = tid; i < kC1 * kW; i += blockDim.x) {
    int c1 = i / kW, w = i % kW;
    int c  = c1 >> 1;
    const float* xp = xb + (size_t)c * kHW + (size_t)hh * kW + w;
    float acc = o1[c1];
    if (hh > 0)      acc += wd1f[c1 * 3 + 0] * xp[-kW];
    acc                  += wd1f[c1 * 3 + 1] * xp[0];
    if (hh < kH - 1) acc += wd1f[c1 * 3 + 2] * xp[kW];
    t1[c1 * kW + w] = (_Float16)acc;
  }
  __syncthreads();

  // ---- stage 2: act = shuffle6(relu6(bn2(dw2(t1)))) along W (1x3, pad 1) -
  for (int i = tid; i < kC2 * kW; i += blockDim.x) {
    int c2 = i / kW, w = i % kW;
    int c1 = c2 >> 1;
    const _Float16* tp = t1 + c1 * kW;
    float acc = o2[c2];
    if (w > 0)      acc += wd2f[c2 * 3 + 0] * (float)tp[w - 1];
    acc                 += wd2f[c2 * 3 + 1] * (float)tp[w];
    if (w < kW - 1) acc += wd2f[c2 * 3 + 2] * (float)tp[w + 1];
    acc = fminf(fmaxf(acc, 0.0f), 6.0f);
    int j = (c2 & 63) * 6 + (c2 >> 6);          // channel-shuffle(groups=6)
    act[w * kKS + j] = (_Float16)acc;
  }
  __syncthreads();

  // ---- stage 3: GEMM  h[96 x 112] = Wp[96 x 384] * act[384 x 112] --------
  const int wave = tid >> 5;
  const int lane = tid & 31;
  if (wave < 7) {                                // 7 column tiles of 16 (wave-uniform branch)
    const int n  = lane & 15;
    const int lo = lane >> 4;
    // B-operand (32x16 f16) per-lane layout, ISA 7.12.4 analogue:
    //   lanes 0-15 hold K={0..7,16..23}, lanes 16-31 hold K={8..15,24..31}
    const _Float16* bcol = act + (wave * 16 + n) * kKS + 8 * lo;

    v8f acc[6];
    v8f zero = {};
#pragma unroll
    for (int mt = 0; mt < 6; ++mt) acc[mt] = zero;

    for (int ks = 0; ks < 12; ++ks) {
      const _Float16* bp = bcol + ks * 32;
      v8h blo = *(const v8h*)(bp);
      v8h bhi = *(const v8h*)(bp + 16);
      v16h bv;
#pragma unroll
      for (int q = 0; q < 8; ++q) { bv[q] = blo[q]; bv[q + 8] = bhi[q]; }
#pragma unroll
      for (int mt = 0; mt < 6; ++mt) {
        v16h av = *(const v16h*)(wp + ((mt * 12 + ks) * 32 + lane) * 16);
        acc[mt] = __builtin_amdgcn_wmma_f32_16x16x32_f16(
            false, av, false, bv, (short)0, acc[mt], false, false);
      }
    }

    // epilogue: bn3 + store h + spatial-sum accumulation
    size_t obase = ((size_t)b * kC) * kHW + (size_t)hh * kW + wave * 16 + n;
#pragma unroll
    for (int mt = 0; mt < 6; ++mt) {
#pragma unroll
      for (int r = 0; r < 8; ++r) {
        int co = mt * 16 + r + 8 * lo;           // C/D layout: VGPR r -> M=r (lanes<16) / r+8
        float v = acc[mt][r] * s3[co] + o3[co];
        out[obase + (size_t)co * kHW] = v;
        float s = v;                              // reduce over the 16 n-lanes of this half
        s += __shfl_xor(s, 1);
        s += __shfl_xor(s, 2);
        s += __shfl_xor(s, 4);
        s += __shfl_xor(s, 8);
        if (n == 0) atomicAdd(&bsm[co], s);
      }
    }
  }
  __syncthreads();
  if (tid < kC) atomicAdd(&gsums[b * kC + tid], bsm[tid]);
}

// ---------------------------------------------------------------------------
// Kernel 2: SE squeeze MLP -> a1/b1 per (batch, channel)
// ---------------------------------------------------------------------------
__global__ void squeeze_kernel(const float* __restrict__ fc1w, const float* __restrict__ fc1b,
                               const float* __restrict__ fc2w, const float* __restrict__ fc2b,
                               char* __restrict__ ws) {
  __shared__ float y[kB * kC];
  __shared__ float z[kB * kSQ];
  const float* sums = (const float*)(ws + SUM_OFF);
  float* a1 = (float*)(ws + A1_OFF);
  float* b1 = (float*)(ws + B1_OFF);
  const int tid = threadIdx.x;

  for (int i = tid; i < kB * kC; i += blockDim.x) y[i] = sums[i] * (1.0f / (float)kHW);
  __syncthreads();
  for (int i = tid; i < kB * kSQ; i += blockDim.x) {
    int b = i / kSQ, s = i % kSQ;
    float acc = fc1b[s];
    for (int c = 0; c < kC; ++c) acc += fc1w[s * kC + c] * y[b * kC + c];
    z[i] = fmaxf(acc, 0.0f);
  }
  __syncthreads();
  for (int i = tid; i < kB * 2 * kC; i += blockDim.x) {
    int b = i / (2 * kC), o = i % (2 * kC);
    float acc = fc2b[o];
    for (int s = 0; s < kSQ; ++s) acc += fc2w[o * kSQ + s] * z[b * kSQ + s];
    acc = fminf(fmaxf(acc + 3.0f, 0.0f), 6.0f) * (1.0f / 6.0f);
    acc = (acc - 0.5f) * 4.0f;
    if (o < kC) a1[b * kC + o] = acc + 1.0f;
    else        b1[b * kC + (o - kC)] = acc;
  }
}

// ---------------------------------------------------------------------------
// Kernel 3: per-pixel dynamic channel mix (roll -1), shuffle(48), residual.
// In-place on d_out: each pixel's 96 channels owned by exactly one thread.
// ---------------------------------------------------------------------------
__global__ void finalize_kernel(const float* __restrict__ x,
                                float* __restrict__ out,
                                const char* __restrict__ ws) {
  const int idx = blockIdx.x * blockDim.x + threadIdx.x;   // grid covers exactly B*HW
  const int b = idx / kHW;
  const int p = idx % kHW;
  const float* a1 = (const float*)(ws + A1_OFF) + b * kC;
  const float* b1 = (const float*)(ws + B1_OFF) + b * kC;
  const size_t base = (size_t)b * kC * kHW + p;

  float hv[kC];
#pragma unroll
  for (int c = 0; c < kC; ++c) hv[c] = out[base + (size_t)c * kHW];

#pragma unroll
  for (int cp = 0; cp < kC; ++cp) {
    int c = (cp % 48) * 2 + cp / 48;             // inverse of shuffle(groups=48)
    float hs = hv[c] * a1[c] + hv[(c + 1) % kC] * b1[c];
    out[base + (size_t)cp * kHW] = hs + x[base + (size_t)cp * kHW];
  }
}

// ---------------------------------------------------------------------------
extern "C" void kernel_launch(void* const* d_in, const int* in_sizes, int n_in,
                              void* d_out, int out_size, void* d_ws, size_t ws_size,
                              hipStream_t stream) {
  (void)in_sizes; (void)n_in; (void)out_size; (void)ws_size;
  const float* x    = (const float*)d_in[0];
  const float* wdw1 = (const float*)d_in[1];
  const float* g1   = (const float*)d_in[2];
  const float* bb1  = (const float*)d_in[3];
  const float* m1   = (const float*)d_in[4];
  const float* vv1  = (const float*)d_in[5];
  const float* wdw2 = (const float*)d_in[6];
  const float* g2   = (const float*)d_in[7];
  const float* bb2  = (const float*)d_in[8];
  const float* m2   = (const float*)d_in[9];
  const float* vv2  = (const float*)d_in[10];
  const float* wpw  = (const float*)d_in[11];
  const float* g3   = (const float*)d_in[12];
  const float* bb3  = (const float*)d_in[13];
  const float* m3   = (const float*)d_in[14];
  const float* vv3  = (const float*)d_in[15];
  const float* fc1w = (const float*)d_in[16];
  const float* fc1b = (const float*)d_in[17];
  const float* fc2w = (const float*)d_in[18];
  const float* fc2b = (const float*)d_in[19];

  char*  ws  = (char*)d_ws;
  float* out = (float*)d_out;

  prep_kernel<<<64, 256, 0, stream>>>(wdw1, g1, bb1, m1, vv1,
                                      wdw2, g2, bb2, m2, vv2,
                                      wpw, g3, bb3, m3, vv3, ws);
  fused_main_kernel<<<kB * kH, 256, LDS_SIZE, stream>>>(x, out, ws);
  squeeze_kernel<<<1, 256, 0, stream>>>(fc1w, fc1b, fc2w, fc2b, ws);
  finalize_kernel<<<(kB * kHW) / 256, 256, 0, stream>>>(x, out, ws);
}
